// AMPred_MFG_91027536872107
// MI455X (gfx1250) — matrix operand order repaired and checked by reading, not compiled, not run on hardware
//
#include <hip/hip_runtime.h>

// ---------------------------------------------------------------------------
// AMPred_MFG for MI455X (gfx1250, wave32, WMMA). See round-0 analysis:
// dead atom branch / dead FG-MPNN eliminated; transformer collapses to 3 GEMMs
// (seq-len 1 softmax == 1); proj1 consumes concat([ax]*4) via index gather.
// All GEMMs: bf16 WMMA 16x16x32, f32 accumulate, fused bias/BN/ReLU/sigmoid.
// Block tile 128x128x32, 8 waves, wave tile 32x64 (8 v_wmma per wave/stage).
// ---------------------------------------------------------------------------

typedef __attribute__((ext_vector_type(16))) __bf16 bf16x16;
typedef __attribute__((ext_vector_type(8)))  float  floatx8;

__device__ __forceinline__ __bf16 f2bf(float f) {
    unsigned u = __builtin_bit_cast(unsigned, f);
    unsigned short h = (unsigned short)(u >> 16);   // truncate-to-bf16
    return __builtin_bit_cast(__bf16, h);
}
__device__ __forceinline__ unsigned pack_bf2(float lo, float hi) {
    unsigned a = __builtin_bit_cast(unsigned, lo);
    unsigned b = __builtin_bit_cast(unsigned, hi);
    return (a >> 16) | (b & 0xffff0000u);           // [hi.bf16 : lo.bf16]
}
__device__ __forceinline__ float sigmoidf_(float x) { return 1.f / (1.f + __expf(-x)); }

#define BM 128
#define BN 128
#define BK 32
#define LDT  34   // LDS row stride in halfs (pad -> odd dword stride, conflict-free)
#define LDTD 17   // same stride in dwords

// C = epilogue(A[M,K] @ W[N,K]^T).  Requires M%128==0, N%128==0.
// AMODE==1: A is ax[128,6500]; logical k in [0,26000) gathers n=k/260, c=(k%260)%65.
// VEC==1: K%4==0 -> float4 global loads + packed dword LDS stores.
template <int AMODE, int VEC>
__global__ __launch_bounds__(256) void gemm_bf16(
    const float* __restrict__ A, const float* __restrict__ W,
    const float* __restrict__ bias, const float* __restrict__ gam,
    const float* __restrict__ bet, float* __restrict__ out,
    int M, int N, int K, int ldc, int mode)
{
    __shared__ unsigned sAu[BM * LDTD];
    __shared__ unsigned sBu[BN * LDTD];
    __bf16* sA = (__bf16*)sAu;
    __bf16* sB = (__bf16*)sBu;

    const int tid  = threadIdx.x;
    const int lane = tid & 31;
    const int wave = tid >> 5;
    const int bm   = blockIdx.y * BM;
    const int bn   = blockIdx.x * BN;
    const int wm   = (wave >> 1) * 32;     // 0 | 32 | 64 | 96
    const int wn   = (wave & 1) * 64;      // 0 | 64

    floatx8 acc[2][4];
#pragma unroll
    for (int i = 0; i < 2; ++i)
#pragma unroll
        for (int j = 0; j < 4; ++j)
#pragma unroll
            for (int e = 0; e < 8; ++e) acc[i][j][e] = 0.f;

    // staging: thread t covers row t/2 (0..127), halfs [(t&1)*16, +16)
    const int srow = tid >> 1;
    const int scol = (tid & 1) * 16;
    const float* aRow = A + (size_t)(bm + srow) * (AMODE ? 6500u : (unsigned)K);
    const float* bRow = W + (size_t)(bn + srow) * K;
    unsigned* aDst = &sAu[srow * LDTD + (scol >> 1)];
    unsigned* bDst = &sBu[srow * LDTD + (scol >> 1)];

    for (int k0 = 0; k0 < K; k0 += BK) {
        __syncthreads();
        if (k0 + BK <= K) {               // ---- full tile: branch-free staging
            if (AMODE == 1) {
                float v[16];
#pragma unroll
                for (int i = 0; i < 16; ++i) {
                    int k = k0 + scol + i;
                    int n = k / 260;
                    int c = (k - n * 260) % 65;
                    v[i] = aRow[n * 65 + c];
                }
#pragma unroll
                for (int i = 0; i < 8; ++i) aDst[i] = pack_bf2(v[2*i], v[2*i+1]);
            } else if (VEC) {
                const float4* p = (const float4*)(aRow + k0 + scol);
                float4 q0 = p[0], q1 = p[1], q2 = p[2], q3 = p[3];
                aDst[0] = pack_bf2(q0.x, q0.y); aDst[1] = pack_bf2(q0.z, q0.w);
                aDst[2] = pack_bf2(q1.x, q1.y); aDst[3] = pack_bf2(q1.z, q1.w);
                aDst[4] = pack_bf2(q2.x, q2.y); aDst[5] = pack_bf2(q2.z, q2.w);
                aDst[6] = pack_bf2(q3.x, q3.y); aDst[7] = pack_bf2(q3.z, q3.w);
            } else {
                float v[16];
#pragma unroll
                for (int i = 0; i < 16; ++i) v[i] = aRow[k0 + scol + i];
#pragma unroll
                for (int i = 0; i < 8; ++i) aDst[i] = pack_bf2(v[2*i], v[2*i+1]);
            }
            if (k0 + BK < K) __builtin_prefetch(bRow + k0 + BK, 0, 1);
            if (VEC) {
                const float4* p = (const float4*)(bRow + k0 + scol);
                float4 q0 = p[0], q1 = p[1], q2 = p[2], q3 = p[3];
                bDst[0] = pack_bf2(q0.x, q0.y); bDst[1] = pack_bf2(q0.z, q0.w);
                bDst[2] = pack_bf2(q1.x, q1.y); bDst[3] = pack_bf2(q1.z, q1.w);
                bDst[4] = pack_bf2(q2.x, q2.y); bDst[5] = pack_bf2(q2.z, q2.w);
                bDst[6] = pack_bf2(q3.x, q3.y); bDst[7] = pack_bf2(q3.z, q3.w);
            } else {
                float v[16];
#pragma unroll
                for (int i = 0; i < 16; ++i) v[i] = bRow[k0 + scol + i];
#pragma unroll
                for (int i = 0; i < 8; ++i) bDst[i] = pack_bf2(v[2*i], v[2*i+1]);
            }
        } else {                          // ---- tail: clamped loads, select-to-0
#pragma unroll
            for (int i = 0; i < 16; ++i) {
                int k  = k0 + scol + i;
                int kc = k < K ? k : K - 1;
                float va;
                if (AMODE == 1) {
                    int n = kc / 260;
                    int c = (kc - n * 260) % 65;
                    va = aRow[n * 65 + c];
                } else {
                    va = aRow[kc];
                }
                float vb = bRow[kc];
                va = k < K ? va : 0.f;
                vb = k < K ? vb : 0.f;
                sA[srow * LDT + scol + i] = f2bf(va);
                sB[srow * LDT + scol + i] = f2bf(vb);
            }
        }
        __syncthreads();

        // fragments per ISA 16-bit layouts (wave32)
        bf16x16 fa[2], fb[4];
        const int r16 = lane & 15, hi = lane >> 4;
#pragma unroll
        for (int s = 0; s < 2; ++s)
#pragma unroll
            for (int h = 0; h < 16; ++h) {
                int ka = h + (h >= 8 ? 8 : 0) + hi * 8;   // A: K 0-7/16-23 | 8-15/24-31
                fa[s][h] = sA[(wm + s * 16 + r16) * LDT + ka];
            }
#pragma unroll
        for (int j = 0; j < 4; ++j)
#pragma unroll
            for (int h = 0; h < 16; ++h) {
                int kb = h + (h >= 8 ? 8 : 0) + hi * 8;   // B: K 0-7/16-23 | 8-15/24-31
                fb[j][h] = sB[(wn + j * 16 + r16) * LDT + kb];
            }
#pragma unroll
        for (int i = 0; i < 2; ++i)
#pragma unroll
            for (int j = 0; j < 4; ++j)
                acc[i][j] = __builtin_amdgcn_wmma_f32_16x16x32_bf16(
                    false, fa[i], false, fb[j], (short)0, acc[i][j], false, false);
    }

    // Epilogue: +bias ; mode>=1: BN(eval, mean0/var1)+ReLU ; mode==2: +sigmoid
    const float rs = 0.9999950000374997f;   // 1/sqrt(1+1e-5)
    const int cr = lane & 15, chi = lane >> 4;
#pragma unroll
    for (int i = 0; i < 2; ++i)
#pragma unroll
        for (int j = 0; j < 4; ++j) {
            int col = bn + wn + j * 16 + cr;
            float bv = bias ? bias[col] : 0.f;
            float gv = gam ? gam[col] : 1.f;
            float ev = bet ? bet[col] : 0.f;
#pragma unroll
            for (int rr = 0; rr < 8; ++rr) {
                int row = bm + wm + i * 16 + rr + chi * 8;
                float v = acc[i][j][rr] + bv;
                if (mode >= 1) { v = v * rs * gv + ev; v = v > 0.f ? v : 0.f; }
                if (mode == 2) v = sigmoidf_(v);
                out[(size_t)row * ldc + col] = v;
            }
        }
}

// ------------------------- GAT (one workgroup per graph) --------------------
__global__ __launch_bounds__(256) void gat_kernel(
    const float* __restrict__ X, const float* __restrict__ A,
    const float* __restrict__ W, const float* __restrict__ a,
    float* __restrict__ ax)
{
    const int b = blockIdx.x, t = threadIdx.x;
    __shared__ float sWh[100 * 66];
    __shared__ float sAl[100 * 101];
    __shared__ float s1[100], s2[100];
    const float* Xb = X + (size_t)b * 6500;
    const float* Ab = A + (size_t)b * 10000;

    for (int idx = t; idx < 6500; idx += 256) {        // Wh = X @ gat_W
        int i = idx / 65, c = idx - i * 65;
        float acc = 0.f;
        for (int k = 0; k < 65; ++k) acc += Xb[i * 65 + k] * W[k * 65 + c];
        sWh[i * 66 + c] = acc;
    }
    __syncthreads();
    if (t < 100) {                                     // s1 = Wh@a[:65], s2 = Wh@a[65:]
        float a1 = 0.f, a2 = 0.f;
        for (int c = 0; c < 65; ++c) { float w = sWh[t * 66 + c]; a1 += w * a[c]; a2 += w * a[65 + c]; }
        s1[t] = a1; s2[t] = a2;
    }
    __syncthreads();
    if (t < 100) {                                     // masked leaky-relu + row softmax
        float si = s1[t], m = -3.4e38f;
        for (int j = 0; j < 100; ++j) {
            float e = si + s2[j];
            e = e > 0.f ? e : 0.2f * e;
            e = (Ab[t * 100 + j] > 0.f) ? e : -9e15f;
            sAl[t * 101 + j] = e;
            m = fmaxf(m, e);
        }
        float sum = 0.f;
        for (int j = 0; j < 100; ++j) { float ex = __expf(sAl[t * 101 + j] - m); sAl[t * 101 + j] = ex; sum += ex; }
        float inv = 1.f / sum;
        for (int j = 0; j < 100; ++j) sAl[t * 101 + j] *= inv;
    }
    __syncthreads();
    for (int idx = t; idx < 6500; idx += 256) {        // elu(alpha @ Wh)
        int i = idx / 65, c = idx - i * 65;
        float acc = 0.f;
        for (int j = 0; j < 100; ++j) acc += sAl[i * 101 + j] * sWh[j * 66 + c];
        ax[(size_t)b * 6500 + idx] = acc > 0.f ? acc : __expf(acc) - 1.f;
    }
}

// ------------------------- GRU pointwise ------------------------------------
__global__ void gru_kernel(const float* __restrict__ gi, const float* __restrict__ gh,
                           const float* __restrict__ h, float* __restrict__ hn, int rows)
{
    int idx = blockIdx.x * blockDim.x + threadIdx.x;
    if (idx >= rows * 128) return;
    int n = idx >> 7, c = idx & 127;
    const float* a = gi + (size_t)n * 384;
    const float* b = gh + (size_t)n * 384;
    float r  = sigmoidf_(a[c] + b[c]);
    float z  = sigmoidf_(a[128 + c] + b[128 + c]);
    float nn = tanhf(a[256 + c] + r * b[256 + c]);
    hn[idx] = (1.f - z) * nn + z * h[idx];
}

// ------------------------- attentive readout (segments of 8) ----------------
__global__ void readout_kernel(const float* __restrict__ h, const float* __restrict__ mf,
                               const float* __restrict__ attW, const float* __restrict__ attb,
                               float* __restrict__ mread)
{
    const int b = blockIdx.x, t = threadIdx.x;   // 128 threads
    __shared__ float s[8];
    const float* hb = h + (size_t)b * 8 * 128;
    if (t < 8) {
        float acc = 0.f;
        for (int c = 0; c < 128; ++c) acc += hb[t * 128 + c] * attW[c];
        s[t] = acc + attb[0];
    }
    __syncthreads();
    float m = s[0];
    for (int i = 1; i < 8; ++i) m = fmaxf(m, s[i]);
    float al[8], sum = 0.f;
    for (int i = 0; i < 8; ++i) { al[i] = __expf(s[i] - m); sum += al[i]; }
    float inv = 1.f / sum;
    float ws = 0.f, mx = -3.4e38f;
    for (int i = 0; i < 8; ++i) { float hv = hb[i * 128 + t]; ws += al[i] * inv * hv; mx = fmaxf(mx, hv); }
    mread[b * 359 + t]       = ws;
    mread[b * 359 + 128 + t] = mx;
    if (t < 103) mread[b * 359 + 256 + t] = mf[b * 103 + t];
}

// ------------------------- small fusion kernels -----------------------------
__global__ void diff_kernel(float* __restrict__ bin)
{
    int idx = blockIdx.x * blockDim.x + threadIdx.x;
    if (idx >= 128 * 128) return;
    int b = idx >> 7, c = idx & 127;
    bin[b * 384 + 256 + c] = bin[b * 384 + c] - bin[b * 384 + 128 + c];
}

__global__ void zero_dist(float* __restrict__ out) { if (threadIdx.x == 0) out[384] = 0.f; }

__global__ void final_kernel(const float* __restrict__ bin, const float* __restrict__ beta,
                             const float* __restrict__ w, const float* __restrict__ bb,
                             float* __restrict__ out)
{
    const int b = blockIdx.x, c = threadIdx.x;   // 128 threads
    __shared__ float r1[128], r2[128], r3[128];
    float xr = bin[b * 384 + c];
    float mp = bin[b * 384 + 128 + c];
    float bt = beta[b * 128 + c];
    float o  = bt * xr + (1.f - bt) * mp;
    out[385 + b * 128 + c] = o;                  // 'out' tensor
    float wc = w[c];
    r1[c] = o  * wc;
    r2[c] = mp * wc;
    float d = sigmoidf_(xr) - sigmoidf_(mp);
    r3[c] = d * d;
    __syncthreads();
    for (int st = 64; st > 0; st >>= 1) {
        if (c < st) { r1[c] += r1[c + st]; r2[c] += r2[c + st]; r3[c] += r3[c + st]; }
        __syncthreads();
    }
    if (c == 0) {
        float xo = r1[0] + bb[0], mo = r2[0] + bb[0];
        out[b]       = xo;                       // x_r_o
        out[128 + b] = mo;                       // mp_o
        out[256 + b] = sigmoidf_(xo);            // logits
        atomicAdd(out + 384, r3[0] * (1.f / 16384.f));   // dist (mean)
    }
}

// ---------------------------------------------------------------------------
extern "C" void kernel_launch(void* const* d_in, const int* in_sizes, int n_in,
                              void* d_out, int out_size, void* d_ws, size_t ws_size,
                              hipStream_t stream)
{
    (void)in_sizes; (void)n_in; (void)out_size; (void)ws_size;
#define IN_F(i) ((const float*)d_in[(i)])
    // params flattened alphabetically (jax pytree order) starting at index 11;
    // _lbn_p leaves: W, b, be, g  ->  pass (W, bias=b, gamma=g, beta=be)
#define LBN(base) IN_F(base), IN_F((base) + 1), IN_F((base) + 3), IN_F((base) + 2)

    float* ws = (float*)d_ws;
    size_t off = 0;
    auto alloc = [&](size_t n) { float* p = ws + off; off += n; return p; };
    float* ax  = alloc(128 * 6500);
    float* ufA = alloc(1024 * 128);
    float* ufB = alloc(1024 * 128);
    float* t1  = alloc(1024 * 128);
    float* t2  = alloc(1024 * 128);
    float* t3  = alloc(1024 * 128);
    float* gi  = alloc(1024 * 384);
    float* gh  = alloc(1024 * 384);
    float* p1  = alloc(128 * 1024);
    float* r1  = alloc(128 * 1024);
    float* r2  = alloc(128 * 512);
    float* r3  = alloc(128 * 256);
    float* e1  = alloc(128 * 512);
    float* e2  = alloc(128 * 256);
    float* mrd = alloc(128 * 359);
    float* m1  = alloc(128 * 256);
    float* fu1 = alloc(128 * 512);
    float* sk1 = alloc(128 * 256);
    float* bin = alloc(128 * 384);
    float* b1  = alloc(128 * 256);
    float* bta = alloc(128 * 128);

    auto gemm = [&](const float* A, const float* W, const float* bias, const float* g,
                    const float* be, float* out, int M, int N, int K, int ldc, int mode) {
        dim3 grid(N / 128, M / 128);
        if ((K & 3) == 0)
            gemm_bf16<0, 1><<<grid, 256, 0, stream>>>(A, W, bias, g, be, out, M, N, K, ldc, mode);
        else
            gemm_bf16<0, 0><<<grid, 256, 0, stream>>>(A, W, bias, g, be, out, M, N, K, ldc, mode);
    };

    // ---- molecule-graph branch: GAT -> proj1 (gathered) -> proj2 -> fu1[:,384:]
    gat_kernel<<<128, 256, 0, stream>>>(IN_F(9), IN_F(10), IN_F(61), IN_F(62), ax);
    {
        dim3 grid(1024 / 128, 128 / 128);
        gemm_bf16<1, 1><<<grid, 256, 0, stream>>>(ax, LBN(81), p1, 128, 1024, 26000, 1024, 1);
    }
    gemm(p1, LBN(85), fu1 + 384, 128, 128, 1024, 512, 1);

    // ---- FG node encoder: ufnf = lbn(lbn(fnf, menc1), menc2)
    gemm(IN_F(2), LBN(71), t1, 1024, 128, 64, 128, 1);
    gemm(t1, LBN(75), ufA, 1024, 128, 128, 128, 1);

    // ---- 2 message-passing steps (transformer collapses to 3 GEMMs; gru_m)
    for (int step = 0; step < 2; ++step) {
        float* cur = (step == 0) ? ufA : ufB;
        float* nxt = (step == 0) ? ufB : ufA;
        gemm(cur, IN_F(103), IN_F(107), nullptr, nullptr, t1, 1024, 128, 128, 128, 0); // @Wv^T+bv
        gemm(t1,  IN_F(100), nullptr,   nullptr, nullptr, t2, 1024, 128, 128, 128, 0); // @Wl2^T
        gemm(t2,  IN_F(101), IN_F(105), nullptr, nullptr, t3, 1024, 128, 128, 128, 0); // @Wo^T+bo
        gemm(t3,  IN_F(68),  IN_F(70),  nullptr, nullptr, gi, 1024, 384, 128, 384, 0); // x@Wih^T+bih
        gemm(cur, IN_F(67),  IN_F(69),  nullptr, nullptr, gh, 1024, 384, 128, 384, 0); // h@Whh^T+bhh
        gru_kernel<<<(1024 * 128 + 255) / 256, 256, 0, stream>>>(gi, gh, cur, nxt, 1024);
    }

    // ---- readout -> motif_pred -> bin[:,128:]
    readout_kernel<<<128, 128, 0, stream>>>(ufA, IN_F(4), IN_F(21), IN_F(22), mrd);
    gemm(mrd, LBN(33), m1, 128, 256, 359, 256, 1);
    gemm(m1, LBN(37), bin + 128, 128, 128, 256, 384, 1);

    // ---- fingerprint branches -> fu1 columns
    gemm(IN_F(8), LBN(57), r1, 128, 1024, 2048, 1024, 1);        // rdit -> fprdit
    gemm(r1, LBN(41), r2, 128, 512, 1024, 512, 1);
    gemm(r2, LBN(45), r3, 128, 256, 512, 256, 1);
    gemm(r3, LBN(49), fu1 + 256, 128, 128, 256, 512, 1);         // rdit_h
    gemm(IN_F(7), LBN(41), e1, 128, 512, 1024, 512, 1);          // ecfp = fpe(fp)
    gemm(e1, LBN(45), e2, 128, 256, 512, 256, 1);
    gemm(e2, LBN(49), fu1 + 128, 128, 128, 256, 512, 1);
    gemm(IN_F(6), LBN(53), fu1 + 0, 128, 128, 167, 512, 1);      // macc_h

    // ---- skip path -> x_r -> bin[:,0:]
    gemm(fu1, LBN(89), sk1, 128, 256, 512, 256, 1);
    gemm(sk1, LBN(93), bin + 0, 128, 128, 256, 384, 1);

    // ---- beta gate
    diff_kernel<<<(128 * 128 + 255) / 256, 256, 0, stream>>>(bin);
    gemm(bin, LBN(23), b1, 128, 256, 384, 256, 1);
    gemm(b1, LBN(27), bta, 128, 128, 256, 128, 2);               // sigmoid(lbn(..))

    // ---- fuse + heads + dist
    zero_dist<<<1, 32, 0, stream>>>((float*)d_out);
    final_kernel<<<128, 128, 0, stream>>>(bin, bta, IN_F(31), IN_F(32), (float*)d_out);
#undef LBN
#undef IN_F
}
